// GNN_A3TGCN_ea_48653389529156
// MI455X (gfx1250) — compile-verified, matrix operand order checked by reading, not run on hardware
//
#include <hip/hip_runtime.h>
#include <hip/hip_bf16.h>
#include <math.h>

typedef __attribute__((ext_vector_type(16))) _Float16 v16h;
typedef __attribute__((ext_vector_type(8)))  _Float16 v8h;
typedef __attribute__((ext_vector_type(8)))  float    v8f;

#define NN 50000
#define FF 8
#define TT 12
#define HH 64
#define EE 1600000

#define LOG2E  1.44269504f
#define LOG2E2 2.88539008f

// Workspace layout (float offsets)
#define WS_DEG   0
#define WS_DINV  (NN)
#define WS_Y     (2*NN)                 // N x 96 aggregated features
#define WS_PROBS (2*NN + NN*96)         // softmax(att), 12 (+pad)
#define WS_CZ    (WS_PROBS + 16)        // fused z bias, 64
#define WS_CH    (WS_CZ + 64)           // fused h bias, 64
#define WS_MZ    (WS_CH + 64)           // 8x64 f16 (256 float slots)
#define WS_MH    (WS_MZ + 256)          // 8x64 f16

// ---------------------------------------------------------------- init
__global__ void k_init(float* ws) {
  long i = (long)blockIdx.x * blockDim.x + threadIdx.x;
  if (i < (long)NN * 96) ws[WS_Y + i] = 0.0f;
  if (i < NN)            ws[WS_DEG + i] = 1.0f;   // self-loop weight
}

// ------------------------------------------- fold weights: Mz=Wz@Wlz[:H]
__global__ void k_prep(const float* Wz, const float* bz,
                       const float* Wh, const float* bh,
                       const float* Wlz, const float* blz,
                       const float* Wlh, const float* blh,
                       const float* att, float* ws) {
  int h = threadIdx.x;                       // 0..63
  _Float16* Mz = (_Float16*)(ws + WS_MZ);
  _Float16* Mh = (_Float16*)(ws + WS_MH);
  float cz = blz[h], ch = blh[h];
  for (int k = 0; k < HH; ++k) {
    cz += bz[k] * Wlz[k*HH + h];
    ch += bh[k] * Wlh[k*HH + h];
  }
  ws[WS_CZ + h] = cz;
  ws[WS_CH + h] = ch;
  for (int f = 0; f < FF; ++f) {
    float mz = 0.f, mh = 0.f;
    for (int k = 0; k < HH; ++k) {
      mz += Wz[f*HH + k] * Wlz[k*HH + h];
      mh += Wh[f*HH + k] * Wlh[k*HH + h];
    }
    Mz[f*HH + h] = (_Float16)mz;
    Mh[f*HH + h] = (_Float16)mh;
  }
  if (h == 0) {  // softmax(att)
    float m = att[0];
    for (int t = 1; t < TT; ++t) m = fmaxf(m, att[t]);
    float e[TT], s = 0.f;
    for (int t = 0; t < TT; ++t) {
      e[t] = __builtin_amdgcn_exp2f((att[t] - m) * LOG2E);
      s += e[t];
    }
    float inv = 1.f / s;
    for (int t = 0; t < TT; ++t) ws[WS_PROBS + t] = e[t] * inv;
  }
}

// ---------------------------------------------------------------- degree
__global__ void k_deg(const int* ei, const float* ea, float* ws) {
  int e = blockIdx.x * blockDim.x + threadIdx.x;
  if (e < EE) atomicAdd(&ws[WS_DEG + ei[EE + e]], ea[e]);
}

__global__ void k_dinv(float* ws) {
  int n = blockIdx.x * blockDim.x + threadIdx.x;
  if (n < NN) {
    float d = ws[WS_DEG + n];
    ws[WS_DINV + n] = d > 0.f ? rsqrtf(d) : 0.f;
  }
}

// ------------------------------------------ Y = A_offdiag @ X (scatter)
// One wave per edge (edge id is wave-uniform -> scalar loads); each of the
// 32 lanes scatters 3 of the 96 contiguous features with f32 atomics.
__global__ void k_agg(const float* x, const int* ei, const float* ea, float* ws) {
  long tid = (long)blockIdx.x * blockDim.x + threadIdx.x;
  int e    = (int)(tid >> 5);
  int lane = (int)(tid & 31);
  if (e >= EE) return;
  e = __builtin_amdgcn_readfirstlane(e);     // wave-uniform -> SGPR/s_load path
  int s = ei[e], d = ei[EE + e];
  float w = ws[WS_DINV + s] * ea[e] * ws[WS_DINV + d];
  const float* xs = x + (long)s * 96;
  float*       yd = ws + WS_Y + (long)d * 96;
#pragma unroll
  for (int r = 0; r < 3; ++r) {
    int j = lane + 32 * r;
    atomicAdd(&yd[j], w * xs[j]);
  }
}

// ---------------------- main: per 16-node tile, WMMA gate GEMMs + epilogue
__global__ void __launch_bounds__(32)
k_main(const float* x, const float* Wout, const float* bout,
       const float* ws, float* out) {
  // LDS: f16 A-staging in [t][node][f] order so each lane's per-t A fragment
  // is a single aligned ds_load_b128; plus relu(acc) and staged Wout.
  __shared__ _Float16 ah[TT * 16 * FF];      // 3 KB
  __shared__ float    ost[16 * HH];          // 4 KB
  __shared__ float    wsh[HH * TT];          // 3 KB

  int lane = threadIdx.x;
  int base = blockIdx.x * 16;
  const float* Y    = ws + WS_Y;
  const float* dinv = ws + WS_DINV;

  // stage Y + dinv^2 * X (self-loop term), converted to f16, t-major
  for (int i = lane; i < 16*96; i += 32) {
    int nl = i / 96, j = i % 96;
    int f = j / TT, t = j % TT;
    int n = base + nl;
    float dv = dinv[n];
    float v = Y[(long)n*96 + j] + dv*dv * x[(long)n*96 + j];
    ah[t*(16*FF) + nl*FF + f] = (_Float16)v;
  }
  for (int i = lane; i < HH*TT; i += 32) wsh[i] = Wout[i];
  __syncthreads();

  bool lo  = lane < 16;
  int  col = lane & 15;

  // B fragments: 8x64 f16, K=0..7 in lanes 0..15 (lanes 16..31 = K 8..15 = 0)
  const _Float16* Mz = (const _Float16*)(ws + WS_MZ);
  const _Float16* Mh = (const _Float16*)(ws + WS_MH);
  v16h bz[4], bh[4];
#pragma unroll
  for (int c = 0; c < 4; ++c) {
    v16h tz = {}; v16h th = {};
#pragma unroll
    for (int k = 0; k < 8; ++k) {
      float vz = (float)Mz[k*HH + c*16 + col];
      float vh = (float)Mh[k*HH + c*16 + col];
      tz[k] = (_Float16)(lo ? vz : 0.f);
      th[k] = (_Float16)(lo ? vh : 0.f);
    }
    bz[c] = tz; bh[c] = th;
  }

  // fused biases as C fragments (broadcast over the 8 row-VGPRs)
  v8f czf[4], chf[4];
#pragma unroll
  for (int c = 0; c < 4; ++c) {
    float cz = ws[WS_CZ + c*16 + col];
    float ch = ws[WS_CH + c*16 + col];
    v8f a8, b8;
#pragma unroll
    for (int r = 0; r < 8; ++r) { a8[r] = cz; b8[r] = ch; }
    czf[c] = a8; chf[c] = b8;
  }

  float acc[4][8];
#pragma unroll
  for (int c = 0; c < 4; ++c)
#pragma unroll
    for (int r = 0; r < 8; ++r) acc[c][r] = 0.f;

  for (int t = 0; t < TT; ++t) {
    float p = ws[WS_PROBS + t];
    // A fragment: one b128 LDS load (8 halfs); lanes>=16 carry K=8..15 which
    // multiply the zeroed B rows, so no lane masking needed.
    v8h alo = *(const v8h*)(ah + t*(16*FF) + col*FF);
    v16h a = {};
#pragma unroll
    for (int k = 0; k < 8; ++k) a[k] = alo[k];

#pragma unroll
    for (int c = 0; c < 4; ++c) {
      v8f zd = __builtin_amdgcn_wmma_f32_16x16x32_f16(
          false, a, false, bz[c], (short)0, czf[c], false, false);
      v8f hd = __builtin_amdgcn_wmma_f32_16x16x32_f16(
          false, a, false, bh[c], (short)0, chf[c], false, false);
#pragma unroll
      for (int r = 0; r < 8; ++r) {
        // (1 - sigmoid(z)) = rcp(1 + 2^(z*log2e));  tanh(h) = 1 - 2*rcp(1 + 2^(2h*log2e))
        float ez   = __builtin_amdgcn_exp2f(zd[r] * LOG2E);
        float gate = __builtin_amdgcn_rcpf(1.f + ez);
        float eh   = __builtin_amdgcn_exp2f(hd[r] * LOG2E2);
        float th   = 1.f - 2.f * __builtin_amdgcn_rcpf(1.f + eh);
        acc[c][r] = fmaf(p * gate, th, acc[c][r]);
      }
    }
  }

  // relu(acc) -> LDS (C/D layout: row = r + 8*(lane>=16), col = 16c + lane&15)
  int mrow = (lane >> 4) * 8;
#pragma unroll
  for (int c = 0; c < 4; ++c)
#pragma unroll
    for (int r = 0; r < 8; ++r)
      ost[(mrow + r)*HH + c*16 + col] = fmaxf(acc[c][r], 0.f);
  __syncthreads();

  // out = relu(acc) @ Wout + bout   (16 nodes x 12 outputs, K=64, all in LDS)
  for (int i = lane; i < 16*TT; i += 32) {
    int nl = i / TT, tt = i % TT;
    float s = bout[tt];
#pragma unroll 8
    for (int h = 0; h < HH; ++h) s = fmaf(ost[nl*HH + h], wsh[h*TT + tt], s);
    out[(long)(base + nl)*TT + tt] = s;
  }
}

extern "C" void kernel_launch(void* const* d_in, const int* in_sizes, int n_in,
                              void* d_out, int out_size, void* d_ws, size_t ws_size,
                              hipStream_t stream) {
  (void)in_sizes; (void)n_in; (void)out_size; (void)ws_size;
  const float* x    = (const float*)d_in[0];
  const int*   ei   = (const int*)  d_in[1];
  const float* ea   = (const float*)d_in[2];
  const float* Wz   = (const float*)d_in[3];
  const float* bz   = (const float*)d_in[4];
  // d_in[5]=Wr, d_in[6]=br are dead (H0 stays zero in the reference)
  const float* Wh   = (const float*)d_in[7];
  const float* bh   = (const float*)d_in[8];
  const float* Wlz  = (const float*)d_in[9];
  const float* blz  = (const float*)d_in[10];
  // d_in[11]=Wlr, d_in[12]=blr dead
  const float* Wlh  = (const float*)d_in[13];
  const float* blh  = (const float*)d_in[14];
  const float* att  = (const float*)d_in[15];
  const float* Wout = (const float*)d_in[16];
  const float* bout = (const float*)d_in[17];
  float* ws  = (float*)d_ws;
  float* out = (float*)d_out;

  k_init<<<(NN*96 + 255)/256, 256, 0, stream>>>(ws);
  k_prep<<<1, 64, 0, stream>>>(Wz, bz, Wh, bh, Wlz, blz, Wlh, blh, att, ws);
  k_deg<<<(EE + 255)/256, 256, 0, stream>>>(ei, ea, ws);
  k_dinv<<<(NN + 255)/256, 256, 0, stream>>>(ws);
  long aggT = (long)EE * 32;
  k_agg<<<(int)((aggT + 255)/256), 256, 0, stream>>>(x, ei, ea, ws);
  k_main<<<NN/16, 32, 0, stream>>>(x, Wout, bout, ws, out);
}
